// SIA_60395830117245
// MI455X (gfx1250) — matrix operand order repaired and checked by reading, not compiled, not run on hardware
//
#include <hip/hip_runtime.h>

// ---------------------------------------------------------------------------
// Types for CDNA5 WMMA (wave32): v_wmma_f32_16x16x32_bf16
// ---------------------------------------------------------------------------
typedef __attribute__((ext_vector_type(16))) __bf16 bf16x16;
typedef __attribute__((ext_vector_type(8)))  float  floatx8;

// TDM descriptor groups (6-arg builtin signature on this toolchain:
//   (uint32x4 g0, int32x8 g1, int32x4, int32x4, int32x8, i32 cpol))
typedef __attribute__((ext_vector_type(4))) unsigned int tdm_g0_t;
typedef __attribute__((ext_vector_type(8))) int          tdm_g1_t;
typedef __attribute__((ext_vector_type(4))) int          tdm_g2_t;
typedef __attribute__((ext_vector_type(8))) int          tdm_g4_t;

#define DIMK 1024

__device__ __forceinline__ unsigned short f2bf(float f) {
    unsigned int u = __float_as_uint(f);
    u += 0x7FFFu + ((u >> 16) & 1u);          // round-to-nearest-even
    return (unsigned short)(u >> 16);
}
__device__ __forceinline__ float bf2f(unsigned short h) {
    return __uint_as_float(((unsigned int)h) << 16);
}

// ---------------------------------------------------------------------------
// Small elementwise kernels
// ---------------------------------------------------------------------------
__global__ void k_zero_f32(float* p, long long n) {
    long long i = (long long)blockIdx.x * blockDim.x + threadIdx.x;
    if (i < n) p[i] = 0.0f;
}

__global__ void k_cvt_f32_bf16(const float* __restrict__ in,
                               unsigned short* __restrict__ out, long long n) {
    long long i = (long long)blockIdx.x * blockDim.x + threadIdx.x;
    if (i < n) out[i] = f2bf(in[i]);
}

__global__ void k_edge_deg(const int* __restrict__ src, const int* __restrict__ dst,
                           float* degO, float* degI, int E) {
    int e = blockIdx.x * blockDim.x + threadIdx.x;
    if (e < E) {
        atomicAdd(&degO[src[e]], 1.0f);
        atomicAdd(&degI[dst[e]], 1.0f);
    }
}

__global__ void k_norms(const float* __restrict__ degO, const float* __restrict__ degI,
                        float* __restrict__ nS, float* __restrict__ nD, int N) {
    int i = blockIdx.x * blockDim.x + threadIdx.x;
    if (i < N) {
        float o = degO[i], d = degI[i];
        nS[i] = o > 0.0f ? rsqrtf(fmaxf(o, 1.0f)) : 0.0f;
        nD[i] = d > 0.0f ? rsqrtf(fmaxf(d, 1.0f)) : 0.0f;
    }
}

// pos_hidden = relu(clip(boxes) @ w_pos1 + b_pos1)  -> bf16  (K = 4, VALU-bound)
__global__ void k_pos1(const float* __restrict__ boxes, const float* __restrict__ w1,
                       const float* __restrict__ b1, unsigned short* __restrict__ out, int N) {
    long long idx = (long long)blockIdx.x * blockDim.x + threadIdx.x;
    if (idx >= (long long)N * DIMK) return;
    int n = (int)(idx >> 10), d = (int)(idx & (DIMK - 1));
    float acc = b1[d];
#pragma unroll
    for (int k = 0; k < 4; k++) {
        float b = boxes[n * 4 + k];
        b = fminf(fmaxf(b, -10.0f), 10.0f);
        acc += b * w1[k * DIMK + d];
    }
    out[idx] = f2bf(fmaxf(acc, 0.0f));
}

// out_bf16[n][d] = bf16(agg_f32[n][d] * norm_dst[n])
__global__ void k_scale_cvt(const float* __restrict__ agg, const float* __restrict__ nD,
                            unsigned short* __restrict__ out, int N) {
    long long idx = (long long)blockIdx.x * blockDim.x + threadIdx.x;
    if (idx >= (long long)N * DIMK) return;
    int n = (int)(idx >> 10);
    out[idx] = f2bf(agg[idx] * nD[n]);
}

// agg[dst[e]][:] += h[src[e]][:]   (one block per edge, fp32 atomics)
__global__ __launch_bounds__(256) void k_scatter(const unsigned short* __restrict__ h,
                                                 const int* __restrict__ src,
                                                 const int* __restrict__ dst,
                                                 float* __restrict__ agg, int E) {
    int e = blockIdx.x;
    if (e >= E) return;
    int s = src[e], d = dst[e];
    const unsigned short* hr = h + (long long)s * DIMK;
    float* ar = agg + (long long)d * DIMK;
    int j = threadIdx.x * 4;                       // 256 threads * 4 = 1024
    uint2 pk = *(const uint2*)(hr + j);
    atomicAdd(&ar[j + 0], bf2f((unsigned short)(pk.x & 0xFFFFu)));
    atomicAdd(&ar[j + 1], bf2f((unsigned short)(pk.x >> 16)));
    atomicAdd(&ar[j + 2], bf2f((unsigned short)(pk.y & 0xFFFFu)));
    atomicAdd(&ar[j + 3], bf2f((unsigned short)(pk.y >> 16)));
}

__global__ void k_colsum(const float* __restrict__ h, float* __restrict__ accum,
                         int M, int rowsPer) {
    int d = blockIdx.x * blockDim.x + threadIdx.x;   // 0..1023
    int r0 = blockIdx.y * rowsPer;
    int r1 = r0 + rowsPer; if (r1 > M) r1 = M;
    float s = 0.0f;
    for (int r = r0; r < r1; r++) s += h[(long long)r * DIMK + d];
    atomicAdd(&accum[d], s);
}

__global__ void k_final(const float* __restrict__ accum, float* __restrict__ out, float invM) {
    int d = blockIdx.x * blockDim.x + threadIdx.x;
    if (d < DIMK) out[d] = accum[d] * invM;
}

// ---------------------------------------------------------------------------
// Tiled bf16 WMMA GEMM:  C[M x 1024] = A[M x 1024] @ W[1024 x 1024] + epilogue
//   block = 256 threads (8 waves); block tile 128 x 64; wave tile 32 x 32
//   A tile staged by the Tensor Data Mover (tensor_load_to_lds) with HW row
//   padding; W tile staged transposed via ds_store_b16.
//   mode 0: out_bf = bf16((acc + bias + feats) * scale[row])
//   mode 1: out_bf = bf16(relu(acc + bias) * scale[row])
//   mode 2: out_f  = acc + bias
// ---------------------------------------------------------------------------
#define BM 128
#define BN 64
#define BK 32
#define LDA 40   // ushort stride (80 B) == 16 dwords data + 4 dwords pad (TDM pad)
#define LDB 40

__global__ __launch_bounds__(256) void k_gemm_bf16(
    const unsigned short* __restrict__ A,
    const unsigned short* __restrict__ W,
    const float* __restrict__ bias,
    const float* __restrict__ feats,
    const float* __restrict__ scale,
    unsigned short* __restrict__ outBf,
    float* __restrict__ outF,
    int M, int mode)
{
    __shared__ unsigned short As[BM * LDA];   // 10 KB (TDM destination)
    __shared__ unsigned short Bs[BN * LDB];   //  5 KB (W tile, transposed: [n][k])

    const int tid    = threadIdx.x;
    const int lane   = tid & 31;
    const int wave   = tid >> 5;
    const int wr     = wave & 3;      // wave row group (32 rows)
    const int wc     = wave >> 2;     // wave col group (32 cols)
    const int l16    = lane & 15;
    const int halfId = lane >> 4;
    const int blockRow = blockIdx.y * BM;
    const int blockCol = blockIdx.x * BN;

    floatx8 acc[2][2] = {};

    const int wk = tid >> 3;             // 0..31
    const int wn = (tid & 7) * 8;        // 0..56

    // ---- TDM descriptor (constant part). D# group 1:
    //  data_size=1 (2B), pad_enable=1, pad_interval=3 (16 dwords), pad_amount=3 (4 dwords)
    //  tensor_dim0=32, tensor_dim1=M-blockRow (OOB rows read as zero -> zero fill)
    //  tile_dim0=32, tile_dim1=128, tensor_dim0_stride=1024
    const unsigned rowsRem = (unsigned)(M - blockRow);
    tdm_g1_t g1;
    g1[0] = (1 << 16) | (1 << 20) | (3 << 22) | (3 << 25);
    g1[1] = (int)(32u << 16);                               // tensor_dim0[15:0]
    g1[2] = (int)((rowsRem & 0xFFFFu) << 16);               // dim0 hi=0 | tensor_dim1 lo
    g1[3] = (int)(((rowsRem >> 16) & 0xFFFFu) | (32u << 16)); // tensor_dim1 hi | tile_dim0
    g1[4] = (int)128u;                                      // tile_dim1=128, tile_dim2=0
    g1[5] = (int)1024u;                                     // tensor_dim0_stride lo32
    g1[6] = 0;                                              // stride hi | dim1_stride lo
    g1[7] = 0;
    const tdm_g2_t gz4 = {0, 0, 0, 0};
    const tdm_g4_t gz8 = {0, 0, 0, 0, 0, 0, 0, 0};
    // LDS byte offset of As (flat LDS address truncates to LDS offset)
    const unsigned asBase = (unsigned)(unsigned long long)(const void*)&As[0];

    for (int kk = 0; kk < DIMK; kk += BK) {
        __syncthreads();
        // ---- stage A tile (128 x 32 bf16) via Tensor Data Mover, one issue/WG ----
        if (wave == 0) {
            unsigned long long ga =
                (unsigned long long)(const void*)(A + (long long)blockRow * DIMK + kk);
            tdm_g0_t g0;
            g0.x = 1u;                                       // count=1, user descriptor
            g0.y = asBase;                                   // lds_addr
            g0.z = (unsigned)(ga & 0xFFFFFFFFull);           // global_addr lo
            g0.w = (unsigned)((ga >> 32) & 0x1FFFFFFull) | (2u << 30); // hi | type=2
            __builtin_amdgcn_tensor_load_to_lds(g0, g1, gz4, gz4, gz8, 0);
        }
        // ---- stage W tile transposed (Bs[n][k]) so K-pairs are contiguous ----
        {
            const uint4 wv = *(const uint4*)(W + (long long)(kk + wk) * DIMK + blockCol + wn);
            const unsigned short* w16 = (const unsigned short*)&wv;
#pragma unroll
            for (int i = 0; i < 8; i++)
                Bs[(wn + i) * LDB + wk] = w16[i];
        }
        __builtin_amdgcn_s_wait_tensorcnt((short)0);   // wave0: DMA done; others: no-op
        __syncthreads();

        // ---- build fragments per ISA 16-bit layouts ----
        union Frag { bf16x16 v; unsigned int u[8]; };
        Frag a[2], b[2];
#pragma unroll
        for (int ti = 0; ti < 2; ti++) {
            const unsigned int* ap =
                (const unsigned int*)(As + (wr * 32 + ti * 16 + l16) * LDA);
#pragma unroll
            for (int r = 0; r < 8; r++) {
                // A 16x32 bf16: VGPR r holds K = base(r) + 8*half, base = 2r (r<4), 2r+8 (r>=4)
                int kb = (r < 4 ? 2 * r : 2 * r + 8) + 8 * halfId;
                a[ti].u[r] = ap[kb >> 1];
            }
        }
#pragma unroll
        for (int tj = 0; tj < 2; tj++) {
            const unsigned int* bp =
                (const unsigned int*)(Bs + (wc * 32 + tj * 16 + l16) * LDB);
#pragma unroll
            for (int r = 0; r < 8; r++) {
                // B 32x16 bf16: VGPR r holds K = base(r) + 8*half, base = 2r (r<4), 2r+8 (r>=4)
                int kb = (r < 4 ? 2 * r : 2 * r + 8) + 8 * halfId;
                b[tj].u[r] = bp[kb >> 1];
            }
        }

#pragma unroll
        for (int ti = 0; ti < 2; ti++)
#pragma unroll
            for (int tj = 0; tj < 2; tj++)
                acc[ti][tj] = __builtin_amdgcn_wmma_f32_16x16x32_bf16(
                    false, a[ti].v, false, b[tj].v,
                    (short)0, acc[ti][tj], false, false);
    }

    // ---- epilogue: C layout M = r + 8*half, N = lane%16 ----
#pragma unroll
    for (int ti = 0; ti < 2; ti++) {
#pragma unroll
        for (int tj = 0; tj < 2; tj++) {
            int colg = blockCol + wc * 32 + tj * 16 + l16;
            float bv = bias[colg];
#pragma unroll
            for (int r = 0; r < 8; r++) {
                int rowg = blockRow + wr * 32 + ti * 16 + r + 8 * halfId;
                if (rowg < M) {
                    float v = acc[ti][tj][r] + bv;
                    long long off = (long long)rowg * DIMK + colg;
                    if (mode == 0) {
                        v = (v + feats[off]) * scale[rowg];
                        outBf[off] = f2bf(v);
                    } else if (mode == 1) {
                        v = fmaxf(v, 0.0f) * scale[rowg];
                        outBf[off] = f2bf(v);
                    } else {
                        outF[off] = v;
                    }
                }
            }
        }
    }
}

// ---------------------------------------------------------------------------
// Launcher
// ---------------------------------------------------------------------------
extern "C" void kernel_launch(void* const* d_in, const int* in_sizes, int n_in,
                              void* d_out, int out_size, void* d_ws, size_t ws_size,
                              hipStream_t stream) {
    const float* feats  = (const float*)d_in[0];
    const float* boxes  = (const float*)d_in[1];
    const int*   esrc   = (const int*)d_in[2];
    const int*   edst   = (const int*)d_in[3];
    const float* w_pos1 = (const float*)d_in[4];
    const float* b_pos1 = (const float*)d_in[5];
    const float* w_pos2 = (const float*)d_in[6];
    const float* b_pos2 = (const float*)d_in[7];
    const float* w_g1   = (const float*)d_in[8];
    const float* b_g1   = (const float*)d_in[9];
    const float* w_g2   = (const float*)d_in[10];
    const float* b_g2   = (const float*)d_in[11];
    (void)n_in; (void)out_size; (void)ws_size;

    const int N = in_sizes[0] / DIMK;     // 20000
    const int E = in_sizes[2];            // 180000 (with self loops)
    const long long ND = (long long)N * DIMK;
    const long long WW = (long long)DIMK * DIMK;

    char* ws = (char*)d_ws;
    size_t off = 0;
    auto take = [&](size_t bytes) -> char* {
        char* r = ws + off;
        off = (off + bytes + 255) & ~(size_t)255;
        return r;
    };
    unsigned short* Abf  = (unsigned short*)take((size_t)ND * 2);
    unsigned short* Bbf  = (unsigned short*)take((size_t)ND * 2);
    float*          agg  = (float*)take((size_t)ND * 4);          // also final h2 f32
    unsigned short* wbf0 = (unsigned short*)take((size_t)WW * 2);
    unsigned short* wbf1 = (unsigned short*)take((size_t)WW * 2);
    unsigned short* wbf2 = (unsigned short*)take((size_t)WW * 2);
    float* degO  = (float*)take((size_t)2 * N * 4);  float* degI = degO + N;
    float* nS    = (float*)take((size_t)2 * N * 4);  float* nD   = nS + N;
    float* accum = (float*)take((size_t)DIMK * 4);

    dim3 b256(256);
    dim3 gg(DIMK / BN, (N + BM - 1) / BM);

    // weights -> bf16
    k_cvt_f32_bf16<<<(int)((WW + 255) / 256), b256, 0, stream>>>(w_pos2, wbf0, WW);
    k_cvt_f32_bf16<<<(int)((WW + 255) / 256), b256, 0, stream>>>(w_g1,   wbf1, WW);
    k_cvt_f32_bf16<<<(int)((WW + 255) / 256), b256, 0, stream>>>(w_g2,   wbf2, WW);

    // degrees and norms
    k_zero_f32<<<(int)((2LL * N + 255) / 256), b256, 0, stream>>>(degO, 2LL * N);
    k_edge_deg<<<(E + 255) / 256, b256, 0, stream>>>(esrc, edst, degO, degI, E);
    k_norms<<<(N + 255) / 256, b256, 0, stream>>>(degO, degI, nS, nD, N);

    // pos MLP layer 1 -> Abf (bf16)
    k_pos1<<<(int)((ND + 255) / 256), b256, 0, stream>>>(boxes, w_pos1, b_pos1, Abf, N);

    // h0 = (pos_hidden @ w_pos2 + b_pos2 + feats) * norm_src  -> Bbf
    k_gemm_bf16<<<gg, b256, 0, stream>>>(Abf, wbf0, b_pos2, feats, nS, Bbf, nullptr, N, 0);

    // graph conv 1: scatter, scale, GEMM(+relu, *norm_src)
    k_zero_f32<<<(int)((ND + 255) / 256), b256, 0, stream>>>(agg, ND);
    k_scatter<<<E, b256, 0, stream>>>(Bbf, esrc, edst, agg, E);
    k_scale_cvt<<<(int)((ND + 255) / 256), b256, 0, stream>>>(agg, nD, Abf, N);
    k_gemm_bf16<<<gg, b256, 0, stream>>>(Abf, wbf1, b_g1, nullptr, nS, Bbf, nullptr, N, 1);

    // graph conv 2: scatter, scale, final GEMM -> f32 (into agg)
    k_zero_f32<<<(int)((ND + 255) / 256), b256, 0, stream>>>(agg, ND);
    k_scatter<<<E, b256, 0, stream>>>(Bbf, esrc, edst, agg, E);
    k_scale_cvt<<<(int)((ND + 255) / 256), b256, 0, stream>>>(agg, nD, Abf, N);
    k_gemm_bf16<<<gg, b256, 0, stream>>>(Abf, wbf2, b_g2, nullptr, nullptr, nullptr, agg, N, 2);

    // mean over rows -> d_out
    k_zero_f32<<<(DIMK + 255) / 256, b256, 0, stream>>>(accum, DIMK);
    const int rowsPer = 512;
    dim3 gs(DIMK / 256, (N + rowsPer - 1) / rowsPer);
    k_colsum<<<gs, b256, 0, stream>>>(agg, accum, N, rowsPer);
    k_final<<<(DIMK + 255) / 256, b256, 0, stream>>>(accum, (float*)d_out, 1.0f / (float)N);
}
